// DilatedResBlock_34608846471729
// MI455X (gfx1250) — matrix operand order repaired and checked by reading, not compiled, not run on hardware
//
#include <hip/hip_runtime.h>
#include <hip/hip_bf16.h>

// ---------------------------------------------------------------------------
// Problem constants (from reference)
// ---------------------------------------------------------------------------
#define BB   2
#define NN   8192
#define K1   17
#define CH   53      // rppe channels (padded to 64)

typedef __attribute__((ext_vector_type(16))) __bf16 v16bf;
typedef __attribute__((ext_vector_type(8)))  float  v8f;
typedef unsigned int v4u __attribute__((ext_vector_type(4)));
typedef int          v8i __attribute__((ext_vector_type(8)));
typedef int          v4i __attribute__((ext_vector_type(4)));

union BF16x16 { v16bf v; __bf16 h[16]; };

#if defined(__has_builtin)
#if __has_builtin(__builtin_amdgcn_tensor_load_to_lds) && __has_builtin(__builtin_amdgcn_s_wait_tensorcnt)
#define HAVE_TDM 1
#endif
#endif

// ---------------------------------------------------------------------------
// KNN: per-thread query, LDS point tiles, top-17 insertion sort in registers
// ---------------------------------------------------------------------------
__global__ __launch_bounds__(256)
void knn_kernel(const float* __restrict__ pc, int* __restrict__ idx, int npts) {
    const int b = blockIdx.y;
    const int n = blockIdx.x * 256 + threadIdx.x;
    const float2* P = (const float2*)(pc + (size_t)b * npts * 2);
    __shared__ float2 tile[256];

    float qx = 0.f, qy = 0.f;
    if (n < npts) { float2 q = P[n]; qx = q.x; qy = q.y; }

    float bd[K1]; int bi[K1];
#pragma unroll
    for (int k = 0; k < K1; ++k) { bd[k] = 3.4e38f; bi[k] = 0; }

    for (int m0 = 0; m0 < npts; m0 += 256) {
        __syncthreads();
        tile[threadIdx.x] = P[m0 + threadIdx.x];
        __syncthreads();
        for (int j = 0; j < 256; ++j) {
            float dx = qx - tile[j].x;
            float dy = qy - tile[j].y;
            float d  = dx * dx + dy * dy;
            if (d < bd[K1 - 1]) {
                int pos = K1 - 1;
#pragma unroll
                for (int t = K1 - 2; t >= 0; --t) {
                    if (d < bd[t]) { bd[t + 1] = bd[t]; bi[t + 1] = bi[t]; pos = t; }
                }
                bd[pos] = d; bi[pos] = m0 + j;
            }
        }
    }
    if (n < npts) {
        int* row = idx + ((size_t)b * npts + n) * K1;
#pragma unroll
        for (int k = 0; k < K1; ++k) row[k] = bi[k];
    }
}

// ---------------------------------------------------------------------------
// Geometry: rppe (bf16, padded K=64) + ggf (f32). One block per (b,n).
// rppe[a, 0:2] = p_a ; rppe[a, 2+3i+{0,1}] = p_i - p_a ; rppe[a, 2+3i+2] = |p_a|
// ---------------------------------------------------------------------------
__global__ __launch_bounds__(64)
void geom_kernel(const float* __restrict__ pc, const int* __restrict__ idx,
                 __bf16* __restrict__ rppe, float* __restrict__ ggf, int npts) {
    const int bn = blockIdx.x;                 // 0 .. B*N-1
    const int b  = bn / npts;
    const int t  = threadIdx.x;
    __shared__ float2 pts[K1];

    if (t < K1) {
        int nn = idx[(size_t)bn * K1 + t];
        const float2* P = (const float2*)(pc + (size_t)b * npts * 2);
        pts[t] = P[nn];
    }
    __syncthreads();

    if (t < K1) {
        float2 pa = pts[t];
        float  na = sqrtf(pa.x * pa.x + pa.y * pa.y);
        __bf16* row = rppe + ((size_t)bn * K1 + t) * 64;
        row[0] = (__bf16)pa.x;
        row[1] = (__bf16)pa.y;
#pragma unroll
        for (int i = 0; i < K1; ++i) {
            row[2 + i * 3 + 0] = (__bf16)(pts[i].x - pa.x);
            row[2 + i * 3 + 1] = (__bf16)(pts[i].y - pa.y);
            row[2 + i * 3 + 2] = (__bf16)na;
        }
#pragma unroll
        for (int c = CH; c < 64; ++c) row[c] = (__bf16)0.0f;
    } else if (t == 32) {
        float sx = 0.f, sy = 0.f, sxx = 0.f, syy = 0.f, sxy = 0.f;
#pragma unroll
        for (int i = 0; i < K1; ++i) {
            float x = pts[i].x, y = pts[i].y;
            sx += x; sy += y; sxx += x * x; syy += y * y; sxy += x * y;
        }
        const float inv = 1.0f / (float)K1;
        float mx = sx * inv, my = sy * inv;
        float cov = sxy * inv - mx * my;
        float vx  = sxx * inv - mx * mx;
        float vy  = syy * inv - my * my;
        float m   = cov / (vx + 1e-8f);
        float pd  = 1.0f - cov / sqrtf(vx * vy + 1e-8f);
        ggf[(size_t)bn * 2 + 0] = m;
        ggf[(size_t)bn * 2 + 1] = pd;
    }
}

// ---------------------------------------------------------------------------
// f32 -> bf16 elementwise convert
// ---------------------------------------------------------------------------
__global__ void f2bf_kernel(const float* __restrict__ src, __bf16* __restrict__ dst, size_t n) {
    size_t i = (size_t)blockIdx.x * blockDim.x + threadIdx.x;
    if (i < n) dst[i] = (__bf16)src[i];
}

// ---------------------------------------------------------------------------
// Gather rows: dst[(bn*17+k)*ldd + c] = src[(b*N + idx[row])*C + c]
// ---------------------------------------------------------------------------
__global__ void gather_kernel(const __bf16* __restrict__ src, int C,
                              const int* __restrict__ idx, __bf16* __restrict__ dst,
                              int ldd, int npts, size_t total) {
    size_t i = (size_t)blockIdx.x * blockDim.x + threadIdx.x;
    if (i >= total) return;
    int    c = (int)(i % C);
    size_t r = i / C;                          // 0 .. B*N*17-1
    int    b = (int)(r / ((size_t)npts * K1));
    int   nn = idx[r];
    dst[r * ldd + c] = src[((size_t)b * npts + nn) * C + c];
}

// ---------------------------------------------------------------------------
// WMMA GEMM: out = act(A[M,KPAD](bf16) x W[K,Nout](f32->bf16) + bias)
//   - W staged transposed [Nout][KPAD] in LDS (zero padded rows K..KPAD-1)
//   - A tile (64 rows, contiguous) staged into LDS by the Tensor Data Mover
//     (1D tile descriptor, data_size=2B) when available; vector copy fallback
//   - one wave per 16-row tile, loops over 16-col tiles
//   - OBF/RELU template params -> straight-line epilogue
// ---------------------------------------------------------------------------
template <int KPAD, bool OBF, bool RELU>
__global__ __launch_bounds__(128)
void wmma_gemm_kernel(const __bf16* __restrict__ A, const float* __restrict__ W,
                      const float* __restrict__ bias, void* __restrict__ outp,
                      int K, int Nout, int ldo, int col_off) {
    constexpr int NCH = KPAD / 32;
    __shared__ __align__(16) __bf16 Wlds[128 * KPAD];
    __shared__ __align__(16) __bf16 Alds[64 * KPAD];

    const int row0 = blockIdx.x * 64;

#ifdef HAVE_TDM
    // --- Tensor Data Mover: 1D tile (64*KPAD bf16 elems) global -> LDS -----
    if ((threadIdx.x >> 5) == 0) {
        const unsigned long long gaddr =
            (unsigned long long)(const void*)(A + (size_t)row0 * KPAD);
        const unsigned int ldsaddr = (unsigned int)(unsigned long long)(const void*)&Alds[0];
        const unsigned long long tdim0 = (unsigned long long)gridDim.x * 64ull * (unsigned long long)KPAD;
        const unsigned int tile0 = 64u * (unsigned int)KPAD;   // <= 10240, fits 16 bits

        v4u g0;
        g0[0] = 1u;                                        // count=1, user mode
        g0[1] = ldsaddr;                                   // lds_addr (bytes)
        g0[2] = (unsigned int)(gaddr & 0xFFFFFFFFull);     // global_addr[31:0]
        g0[3] = (unsigned int)((gaddr >> 32) & 0x01FFFFFFull) | (2u << 30); // addr msbs | type=2

        v8i g1;
        g1[0] = 0x10000;                                   // data_size=1 (2 bytes), no flags
        g1[1] = (int)((tdim0 & 0xFFFFull) << 16);          // tensor_dim0[15:0]
        g1[2] = (int)((tdim0 >> 16) & 0xFFFFull);          // tensor_dim0[31:16], tensor_dim1=0
        g1[3] = (int)(tile0 << 16);                        // tile_dim0
        g1[4] = 0;                                         // tile_dim1/2 unused (1D)
        g1[5] = (int)(tdim0 & 0xFFFFFFFFull);              // tensor_dim0_stride lo
        g1[6] = (int)((tdim0 >> 32) & 0xFFFFull);          // tensor_dim0_stride hi
        g1[7] = 0;

        v4i z4 = {};
#if __clang_major__ >= 23
        v8i z8 = {};
        __builtin_amdgcn_tensor_load_to_lds(g0, g1, z4, z4, z8, 0);
#else
        __builtin_amdgcn_tensor_load_to_lds(g0, g1, z4, z4, 0);
#endif
        __builtin_amdgcn_s_wait_tensorcnt(0);
    }
#else
    {
        const uint4* Ag = (const uint4*)(A + (size_t)row0 * KPAD);
        uint4* Al = (uint4*)Alds;
        const int total = 64 * KPAD / 8;
        for (int i = threadIdx.x; i < total; i += 128) Al[i] = Ag[i];
    }
#endif

    // Stage weights, transposed, bf16, zero-padded in K
    for (int i = threadIdx.x; i < Nout * KPAD; i += 128) {
        int nn = i / KPAD, kk = i % KPAD;
        Wlds[i] = (kk < K) ? (__bf16)W[(size_t)kk * Nout + nn] : (__bf16)0.0f;
    }
    __builtin_prefetch(A + ((size_t)row0 + 64) * KPAD, 0, 1);   // global_prefetch_b8
    __syncthreads();

    const int wave  = threadIdx.x >> 5;
    const int lane  = threadIdx.x & 31;
    const int mrow  = lane & 15;
    const int kbA   = (lane >> 4) * 8;    // A: lane<16 -> K 0-7/16-23 ; lane>=16 -> 8-15/24-31
    const int kbB   = (lane >> 4) * 8;    // B: lane<16 -> K 0-7/16-23 ; lane>=16 -> 8-15/24-31
    const int mo    = (lane >> 4) * 8;

    // Preload all A fragments for this wave's 16-row tile
    BF16x16 afrag[NCH];
#pragma unroll
    for (int kc = 0; kc < NCH; ++kc) {
#pragma unroll
        for (int h = 0; h < 16; ++h) {
            int k = kc * 32 + (h >> 3) * 16 + kbA + (h & 7);
            afrag[kc].h[h] = Alds[(wave * 16 + mrow) * KPAD + k];
        }
    }

    for (int nt = 0; nt < (Nout >> 4); ++nt) {
        const int ncol = nt * 16 + (lane & 15);
        v8f acc{};
#pragma unroll
        for (int kc = 0; kc < NCH; ++kc) {
            BF16x16 bfrag;
#pragma unroll
            for (int h = 0; h < 16; ++h)
                bfrag.h[h] = Wlds[ncol * KPAD + kc * 32 + (h >> 3) * 16 + kbB + (h & 7)];
            acc = __builtin_amdgcn_wmma_f32_16x16x32_bf16(
                false, afrag[kc].v, false, bfrag.v, (short)0, acc, false, false);
        }
        const float bv = bias[ncol];
        const size_t base = (size_t)(row0 + wave * 16 + mo) * ldo + col_off + ncol;
        if (OBF) {
            __bf16* po = (__bf16*)outp + base;
#pragma unroll
            for (int r = 0; r < 8; ++r) {
                float v = acc[r] + bv;
                if (RELU) v = fmaxf(v, 0.0f);
                po[(size_t)r * ldo] = (__bf16)v;
            }
        } else {
            float* po = (float*)outp + base;
#pragma unroll
            for (int r = 0; r < 8; ++r) {
                float v = acc[r] + bv;
                if (RELU) v = fmaxf(v, 0.0f);
                po[(size_t)r * ldo] = v;
            }
        }
    }
}

// ---------------------------------------------------------------------------
// Attentive pooling: softmax over channels per neighbour, weighted sum over K1,
// then concat ggf and zero-pad to LDO. One block (C threads) per (b,n).
// ---------------------------------------------------------------------------
__global__ void attpool_kernel(const __bf16* __restrict__ cat,
                               const __bf16* __restrict__ logits,
                               const float* __restrict__ ggf,
                               __bf16* __restrict__ attout, int C, int LDO) {
    const int bn = blockIdx.x;
    const int c  = threadIdx.x;           // blockDim.x == C (power of 2)
    __shared__ float red[128];

    float att = 0.0f;
    for (int k = 0; k < K1; ++k) {
        size_t row = (size_t)bn * K1 + k;
        float  l   = (float)logits[row * C + c];
        red[c] = l; __syncthreads();
        for (int s = C >> 1; s > 0; s >>= 1) {
            if (c < s) red[c] = fmaxf(red[c], red[c + s]);
            __syncthreads();
        }
        float mx = red[0]; __syncthreads();
        float e  = __expf(l - mx);
        red[c] = e; __syncthreads();
        for (int s = C >> 1; s > 0; s >>= 1) {
            if (c < s) red[c] += red[c + s];
            __syncthreads();
        }
        float sum = red[0]; __syncthreads();
        att += (float)cat[row * C + c] * (e / sum);
    }
    attout[(size_t)bn * LDO + c] = (__bf16)att;
    if (c == 0) {
        attout[(size_t)bn * LDO + C]     = (__bf16)ggf[(size_t)bn * 2 + 0];
        attout[(size_t)bn * LDO + C + 1] = (__bf16)ggf[(size_t)bn * 2 + 1];
    }
    for (int p = C + 2 + c; p < LDO; p += C) attout[(size_t)bn * LDO + p] = (__bf16)0.0f;
}

// ---------------------------------------------------------------------------
// Final: out = leaky_relu(x3 + y, 0.2)
// ---------------------------------------------------------------------------
__global__ void final_kernel(const float* __restrict__ x3, const float* __restrict__ y,
                             float* __restrict__ out, size_t n) {
    size_t i = (size_t)blockIdx.x * blockDim.x + threadIdx.x;
    if (i < n) {
        float v = x3[i] + y[i];
        out[i] = v > 0.0f ? v : 0.2f * v;
    }
}

// ---------------------------------------------------------------------------
// Host-side launch helpers
// ---------------------------------------------------------------------------
template <int KPAD>
static void launch_gemm_k(const __bf16* A, const float* W, const float* bias,
                          void* out, int M, int K, int Nout, int ldo, int col_off,
                          bool obf, bool relu, hipStream_t s) {
    dim3 g(M / 64), b(128);
    if (obf) {
        if (relu) wmma_gemm_kernel<KPAD, true,  true ><<<g, b, 0, s>>>(A, W, bias, out, K, Nout, ldo, col_off);
        else      wmma_gemm_kernel<KPAD, true,  false><<<g, b, 0, s>>>(A, W, bias, out, K, Nout, ldo, col_off);
    } else {
        if (relu) wmma_gemm_kernel<KPAD, false, true ><<<g, b, 0, s>>>(A, W, bias, out, K, Nout, ldo, col_off);
        else      wmma_gemm_kernel<KPAD, false, false><<<g, b, 0, s>>>(A, W, bias, out, K, Nout, ldo, col_off);
    }
}

static void launch_gemm(int KPAD, const __bf16* A, const float* W, const float* bias,
                        void* out, int M, int K, int Nout,
                        int ldo, int col_off, bool obf, bool relu, hipStream_t s) {
    switch (KPAD) {
        case 64:  launch_gemm_k<64> (A, W, bias, out, M, K, Nout, ldo, col_off, obf, relu, s); break;
        case 96:  launch_gemm_k<96> (A, W, bias, out, M, K, Nout, ldo, col_off, obf, relu, s); break;
        case 128: launch_gemm_k<128>(A, W, bias, out, M, K, Nout, ldo, col_off, obf, relu, s); break;
        case 160: launch_gemm_k<160>(A, W, bias, out, M, K, Nout, ldo, col_off, obf, relu, s); break;
    }
}

extern "C" void kernel_launch(void* const* d_in, const int* in_sizes, int n_in,
                              void* d_out, int out_size, void* d_ws, size_t ws_size,
                              hipStream_t stream) {
    (void)in_sizes; (void)n_in; (void)out_size; (void)ws_size;

    const float* pc    = (const float*)d_in[0];
    const float* feats = (const float*)d_in[1];
    const float* W_res = (const float*)d_in[2];
    const float* b_res = (const float*)d_in[3];
    const float* W0    = (const float*)d_in[4];
    const float* b0    = (const float*)d_in[5];
    const float* Wl0   = (const float*)d_in[6];
    const float* bl0   = (const float*)d_in[7];
    const float* Ws0   = (const float*)d_in[8];
    const float* bs0   = (const float*)d_in[9];
    const float* Wf0   = (const float*)d_in[10];
    const float* bf0   = (const float*)d_in[11];
    const float* Wl1   = (const float*)d_in[12];
    const float* bl1   = (const float*)d_in[13];
    const float* Ws1   = (const float*)d_in[14];
    const float* bs1   = (const float*)d_in[15];
    const float* Wf1   = (const float*)d_in[16];
    const float* bf1   = (const float*)d_in[17];
    const float* W1    = (const float*)d_in[18];
    const float* b1    = (const float*)d_in[19];

    const size_t BN  = (size_t)BB * NN;        // 16384
    const size_t BNK = BN * K1;                // 278528

    // Workspace carve-out (256B aligned)
    char* ws = (char*)d_ws;
    size_t off = 0;
    auto alloc = [&](size_t bytes) -> void* {
        void* p = ws + off;
        off = (off + bytes + 255) & ~(size_t)255;
        return p;
    };
    int*    idxb    = (int*)   alloc(BNK * sizeof(int));
    __bf16* fbf     = (__bf16*)alloc(BN  * 64  * 2);    // feats bf16
    __bf16* rppe    = (__bf16*)alloc(BNK * 64  * 2);    // padded 53->64
    float*  ggf     = (float*) alloc(BN  * 2   * 4);
    float*  ybuf    = (float*) alloc(BN  * 128 * 4);    // MLP_res output (f32)
    __bf16* x0      = (__bf16*)alloc(BN  * 32  * 2);
    __bf16* cat0    = (__bf16*)alloc(BNK * 64  * 2);    // [nf0 | r0]
    __bf16* logit0  = (__bf16*)alloc(BNK * 64  * 2);
    __bf16* att0    = (__bf16*)alloc(BN  * 96  * 2);    // 66 used, padded to 96
    __bf16* x1      = (__bf16*)alloc(BN  * 64  * 2);
    __bf16* cat1    = (__bf16*)alloc(BNK * 128 * 2);    // [nf1 | r1]
    __bf16* logit1  = (__bf16*)alloc(BNK * 128 * 2);
    __bf16* att1    = (__bf16*)alloc(BN  * 160 * 2);    // 130 used, padded to 160
    __bf16* x2      = (__bf16*)alloc(BN  * 128 * 2);
    float*  x3      = (float*) alloc(BN  * 128 * 4);

    // 1) KNN (shared across both LocSE stages)
    knn_kernel<<<dim3(NN / 256, BB), 256, 0, stream>>>(pc, idxb, NN);
    // 2) Geometry: rppe (bf16, K-pad 64) + ggf
    geom_kernel<<<(int)BN, 64, 0, stream>>>(pc, idxb, rppe, ggf, NN);
    // 3) feats -> bf16
    {
        size_t n = BN * 64;
        f2bf_kernel<<<(int)((n + 255) / 256), 256, 0, stream>>>(feats, fbf, n);
    }
    // 4) y = relu(feats @ W_res + b_res)          (f32 out, for residual)
    launch_gemm(64, fbf, W_res, b_res, ybuf, (int)BN, 64, 128, 128, 0, false, true, stream);
    // 5) x0 = relu(feats @ W0 + b0)               (32 ch)
    launch_gemm(64, fbf, W0, b0, x0, (int)BN, 64, 32, 32, 0, true, true, stream);
    // 6) r0 = relu(rppe @ Wl0 + bl0) -> cat0[:,32:64]
    launch_gemm(64, rppe, Wl0, bl0, cat0, (int)BNK, CH, 32, 64, 32, true, true, stream);
    // 7) nf0 = gather(x0) -> cat0[:,0:32]
    {
        size_t total = BNK * 32;
        gather_kernel<<<(int)((total + 255) / 256), 256, 0, stream>>>(x0, 32, idxb, cat0, 64, NN, total);
    }
    // 8) logits0 = cat0 @ Ws0 + bs0  (no relu)
    launch_gemm(64, cat0, Ws0, bs0, logit0, (int)BNK, 64, 64, 64, 0, true, false, stream);
    // 9) attention pool 0 -> att0 (66 used / 96)
    attpool_kernel<<<(int)BN, 64, 0, stream>>>(cat0, logit0, ggf, att0, 64, 96);
    // 10) x1 = relu(att0 @ Wf0 + bf0)             (64 ch)
    launch_gemm(96, att0, Wf0, bf0, x1, (int)BN, 66, 64, 64, 0, true, true, stream);
    // 11) r1 = relu(rppe @ Wl1 + bl1) -> cat1[:,64:128]
    launch_gemm(64, rppe, Wl1, bl1, cat1, (int)BNK, CH, 64, 128, 64, true, true, stream);
    // 12) nf1 = gather(x1) -> cat1[:,0:64]
    {
        size_t total = BNK * 64;
        gather_kernel<<<(int)((total + 255) / 256), 256, 0, stream>>>(x1, 64, idxb, cat1, 128, NN, total);
    }
    // 13) logits1 = cat1 @ Ws1 + bs1 (no relu)
    launch_gemm(128, cat1, Ws1, bs1, logit1, (int)BNK, 128, 128, 128, 0, true, false, stream);
    // 14) attention pool 1 -> att1 (130 used / 160)
    attpool_kernel<<<(int)BN, 128, 0, stream>>>(cat1, logit1, ggf, att1, 128, 160);
    // 15) x2 = relu(att1 @ Wf1 + bf1)             (128 ch)
    launch_gemm(160, att1, Wf1, bf1, x2, (int)BN, 130, 128, 128, 0, true, true, stream);
    // 16) x3 = relu(x2 @ W1 + b1)                 (f32 out)
    launch_gemm(128, x2, W1, b1, x3, (int)BN, 128, 128, 128, 0, false, true, stream);
    // 17) out = leaky_relu(x3 + y, 0.2)
    {
        size_t n = BN * 128;
        final_kernel<<<(int)((n + 255) / 256), 256, 0, stream>>>(x3, ybuf, (float*)d_out, n);
    }
}
